// WindowAttention_76751065579900
// MI455X (gfx1250) — matrix operand order repaired and checked by reading, not compiled, not run on hardware
//
#include <hip/hip_runtime.h>
#include <hip/hip_bf16.h>

// ---------------------------------------------------------------------------
// MI455X (gfx1250) fused window-attention, bf16 WMMA everywhere:
//   k_prep      : w_qkv / w_proj -> bf16 transposed (K-contiguous) copies
//   k_zero_pads : zero the padded rows/cols of the q/k/v^T workspace
//   k_qkv       : x @ w_qkv (A staged via LDS, B direct from L2-resident
//                 transposed weights), scatter q*scale / k / v^T (padded)
//   k_attn_proj : per-window QK^T -> softmax -> PV -> @w_proj + bias; LDS
//                 staged with gfx1250 async-to-LDS copies (ASYNCcnt)
// ---------------------------------------------------------------------------

typedef __bf16 bf16;
typedef __attribute__((ext_vector_type(16))) __bf16       v16bf;
typedef __attribute__((ext_vector_type(8)))  float        v8f;
typedef __attribute__((ext_vector_type(4)))  unsigned int u32x4;
typedef __attribute__((ext_vector_type(4)))  float        f32x4;

#define NH 6     // heads
#define NT 49    // tokens per window
#define CD 192   // channels
#define HD 32    // head dim
#define NW 64    // mask windows
#define NB 4096  // windows (batch)

union FragU {
  v16bf v;
  struct { u32x4 lo; u32x4 hi; } q;
};

// A-matrix 16x32 bf16 fragment (ISA 7.12.2): lane l holds row m=l&15,
// elems 0..7 = K[kb..kb+7], elems 8..15 = K[16+kb..16+kb+7], kb=8*(l>>4).
__device__ __forceinline__ v16bf load_a_frag(const bf16* base, int ld) {
  int l  = threadIdx.x & 31;
  int m  = l & 15;
  int kb = (l >> 4) * 8;
  FragU f;
  f.q.lo = *(const u32x4*)(base + m * ld + kb);
  f.q.hi = *(const u32x4*)(base + m * ld + kb + 16);
  return f.v;
}

// B-matrix 32x16 bf16 fragment from a TRANSPOSED operand (rows = 16 output
// cols, K contiguous): lane l holds col n=l&15, elems 0..7 = K[kh..kh+7],
// elems 8..15 = K[16+kh..16+kh+7], kh=8*(l>>4).
__device__ __forceinline__ v16bf load_b_frag(const bf16* baseT, int ld) {
  int l  = threadIdx.x & 31;
  int n  = l & 15;
  int kh = (l >> 4) * 8;
  FragU f;
  f.q.lo = *(const u32x4*)(baseT + n * ld + kh);
  f.q.hi = *(const u32x4*)(baseT + n * ld + kh + 16);
  return f.v;
}

__device__ __forceinline__ v8f wmma_bf16(v16bf a, v16bf b, v8f c) {
  return __builtin_amdgcn_wmma_f32_16x16x32_bf16(false, a, false, b,
                                                 (short)0, c, false, false);
}

// gfx1250 async global->LDS copy (16B per lane), tracked by ASYNCcnt.
__device__ __forceinline__ void async_b128(unsigned lds_off, const void* gaddr) {
  asm volatile("global_load_async_to_lds_b128 %0, %1, off"
               :: "v"(lds_off), "v"(gaddr) : "memory");
}
__device__ __forceinline__ void wait_async0() {
  asm volatile("s_wait_asynccnt 0x0" ::: "memory");
}

// ---------------------------------------------------------------------------
// K0: transpose+convert weights once.
//   wqkvT[c][k] = w_qkv[k][c]  (576 x 192 bf16, 221KB, L2-resident)
//   wpt  [n][k] = w_proj[k][n] (192 x 192 bf16)
// ---------------------------------------------------------------------------
__global__ void k_prep(const float* __restrict__ wqkv,
                       const float* __restrict__ wproj,
                       bf16* __restrict__ wqkvT, bf16* __restrict__ wpt) {
  int i = blockIdx.x * blockDim.x + threadIdx.x;
  if (i < 3 * CD * CD) {
    int c = i / CD, k = i - c * CD;
    wqkvT[i] = (bf16)wqkv[k * (3 * CD) + c];
  }
  int j = i - 3 * CD * CD;
  if (j >= 0 && j < CD * CD) {
    int n = j / CD, k = j - n * CD;
    wpt[j] = (bf16)wproj[k * CD + n];
  }
}

// ---------------------------------------------------------------------------
// K0b: zero the pad regions of the padded q/k/v^T workspace so padded V
// columns contribute exact zeros (0 * garbage could be NaN otherwise).
//   Q,K: [b][h][64][32], rows 49..63 -> 480 contiguous elems per (b,h)
//   Vt : [b][h][32][64], cols 48..63 per d-row (col 48 rewritten by k_qkv)
// ---------------------------------------------------------------------------
__global__ void k_zero_pads(bf16* __restrict__ Qw, bf16* __restrict__ Kw,
                            bf16* __restrict__ Vt) {
  const u32x4 z = {};
  int idx = blockIdx.x * blockDim.x + threadIdx.x;
  int stride = gridDim.x * blockDim.x;
  // Q/K pads: (b,h) = 24576 units x 60 16B chunks
  for (int i = idx; i < NB * NH * 60; i += stride) {
    int bh = i / 60, c = i - bh * 60;
    size_t off = (size_t)bh * (64 * HD) + NT * HD + c * 8;
    *(u32x4*)(Qw + off) = z;
    *(u32x4*)(Kw + off) = z;
  }
  // Vt pads: (b,h,d) = 786432 rows x 2 chunks (cols 48..63)
  for (int i = idx; i < NB * NH * HD * 2; i += stride) {
    int u = i >> 1, c = i & 1;
    *(u32x4*)(Vt + (size_t)u * 64 + 48 + c * 8) = z;
  }
}

// ---------------------------------------------------------------------------
// K1: qkv = x @ w_qkv.  64x64 tile per block, K=192 in 6 steps of 32.
// A (x) staged f32->bf16 via LDS; B fragments straight from global wqkvT.
// Epilogue scatters into Q[b][h][64][32] (scaled), K[b][h][64][32],
// Vt[b][h][32][64].
// ---------------------------------------------------------------------------
__global__ void __launch_bounds__(256)
k_qkv(const float* __restrict__ x, const bf16* __restrict__ wqkvT,
      bf16* __restrict__ Qw, bf16* __restrict__ Kw, bf16* __restrict__ Vt) {
  __shared__ __align__(16) bf16 sA[64 * 32];   // x tile, row-major bf16

  const int tid = threadIdx.x;
  const int w   = tid >> 5, l = tid & 31;
  const int hi  = l >> 4, col = l & 15;
  const int row0 = blockIdx.y * 64;
  const int c0   = blockIdx.x * 64;
  const int mt     = w & 3;          // wave's M-tile
  const int ntBase = (w >> 2) * 2;   // wave's two N-tiles

  const int am = tid >> 2, ak = (tid & 3) * 8;    // A loader map: 8 f32 each

  v8f acc0 = {}, acc1 = {};

  for (int ks = 0; ks < 6; ++ks) {
    const int k0 = ks * 32;
    f32x4 a0 = *(const f32x4*)(x + (size_t)(row0 + am) * CD + k0 + ak);
    f32x4 a1 = *(const f32x4*)(x + (size_t)(row0 + am) * CD + k0 + ak + 4);
    __syncthreads();                 // previous iteration's readers done
#pragma unroll
    for (int j = 0; j < 4; ++j) {
      sA[am * 32 + ak + j]     = (bf16)a0[j];
      sA[am * 32 + ak + 4 + j] = (bf16)a1[j];
    }
    __syncthreads();
    v16bf af  = load_a_frag(sA + mt * 16 * 32, 32);
    v16bf bf0 = load_b_frag(wqkvT + (size_t)(c0 + (ntBase + 0) * 16) * CD + k0, CD);
    v16bf bf1 = load_b_frag(wqkvT + (size_t)(c0 + (ntBase + 1) * 16) * CD + k0, CD);
    acc0 = wmma_bf16(af, bf0, acc0);
    acc1 = wmma_bf16(af, bf1, acc1);
  }

  const float qscale = 0.17677669529663687f;  // 32^-0.5
#pragma unroll
  for (int t = 0; t < 2; ++t) {
    const int nt = ntBase + t;
    v8f acc = t ? acc1 : acc0;
#pragma unroll
    for (int r = 0; r < 8; ++r) {
      int gr  = row0 + mt * 16 + hi * 8 + r;   // global row = b*49 + tok
      int c   = c0 + nt * 16 + col;            // 0..575
      int b   = gr / NT;
      int tok = gr - b * NT;
      float v = acc[r];
      if (c < CD) {
        int h = c >> 5, d = c & 31;
        Qw[(((size_t)b * NH + h) * 64 + tok) * HD + d] = (bf16)(v * qscale);
      } else if (c < 2 * CD) {
        int cc = c - CD; int h = cc >> 5, d = cc & 31;
        Kw[(((size_t)b * NH + h) * 64 + tok) * HD + d] = (bf16)v;
      } else {
        int cc = c - 2 * CD; int h = cc >> 5, d = cc & 31;
        Vt[(((size_t)b * NH + h) * HD + d) * 64 + tok] = (bf16)v;  // transposed
      }
    }
  }
}

// ---------------------------------------------------------------------------
// K2: per-window attention + output projection.  8 waves; waves 0..5 own one
// head each (S=QK^T, softmax, P·V); all 8 waves do ctx @ w_proj + bias.
// Staging uses gfx1250 async-to-LDS (3 contiguous 24KB copies per block).
// ---------------------------------------------------------------------------
__global__ void __launch_bounds__(256)
k_attn_proj(const bf16* __restrict__ Qw, const bf16* __restrict__ Kw,
            const bf16* __restrict__ Vt, const float* __restrict__ mask,
            const bf16* __restrict__ wpt, const float* __restrict__ bias,
            float* __restrict__ out) {
  __shared__ __align__(16) bf16 sQ[NH * 64 * 32];  // rows 49..63 are zeros
  __shared__ __align__(16) bf16 sK[NH * 64 * 32];
  __shared__ __align__(16) bf16 sV[NH * 32 * 64];  // [h][d][n], n>=49 zeros
  __shared__ __align__(16) bf16 sP[NH * 64 * 64];  // softmax probs per head
  __shared__ __align__(16) bf16 sC[64 * 192];      // per-window context

  const int b   = blockIdx.x;
  const int tid = threadIdx.x;
  const int w   = tid >> 5, l = tid & 31;
  const int hi  = l >> 4, col = l & 15;

  // ---- async stage q/k/v^T into LDS (16B per lane per op) ----
  {
    const bf16* gQ = Qw + (size_t)b * (NH * 64 * HD);
    const bf16* gK = Kw + (size_t)b * (NH * 64 * HD);
    const bf16* gV = Vt + (size_t)b * (NH * HD * 64);
    for (int i = tid; i < (NH * 64 * HD) / 8; i += 256) {  // 1536 chunks each
      int e = i * 8;
      async_b128((unsigned)(uintptr_t)(sQ + e), gQ + e);
      async_b128((unsigned)(uintptr_t)(sK + e), gK + e);
      async_b128((unsigned)(uintptr_t)(sV + e), gV + e);
    }
    wait_async0();
  }
  __syncthreads();

  const float* mwin = mask + (size_t)(b & (NW - 1)) * NT * NT;

  if (w < NH) {
    const int h = w;
    const bf16* Qh = sQ + h * 2048;
    const bf16* Kh = sK + h * 2048;

    // ---- S = Q K^T : 4x4 tiles of 16x16, K=d=32 (one WMMA each) ----
    v8f accS[4][4] = {};
    v16bf qa[4], kbm[4];
#pragma unroll
    for (int mt = 0; mt < 4; ++mt) qa[mt] = load_a_frag(Qh + mt * 512, 32);
#pragma unroll
    for (int nt = 0; nt < 4; ++nt) kbm[nt] = load_b_frag(Kh + nt * 512, 32);
#pragma unroll
    for (int mt = 0; mt < 4; ++mt)
#pragma unroll
      for (int nt = 0; nt < 4; ++nt)
        accS[mt][nt] = wmma_bf16(qa[mt], kbm[nt], accS[mt][nt]);

    // ---- mask add + row softmax (16-lane shuffle reduce) + P to LDS ----
    bf16* Ph = sP + h * 4096;
#pragma unroll
    for (int mt = 0; mt < 4; ++mt) {
#pragma unroll
      for (int r = 0; r < 8; ++r) {
        int m = mt * 16 + hi * 8 + r;
        float s[4];
#pragma unroll
        for (int nt = 0; nt < 4; ++nt) {
          int n = nt * 16 + col;
          float sv = accS[mt][nt][r];
          if (n < NT) { if (m < NT) sv += mwin[m * NT + n]; }
          else sv = -1e30f;                     // kill padded columns
          s[nt] = sv;
        }
        float mx = fmaxf(fmaxf(s[0], s[1]), fmaxf(s[2], s[3]));
#pragma unroll
        for (int d2 = 1; d2 < 16; d2 <<= 1)
          mx = fmaxf(mx, __shfl_xor(mx, d2, 32));
        float e[4], sum = 0.f;
#pragma unroll
        for (int nt = 0; nt < 4; ++nt) { e[nt] = __expf(s[nt] - mx); sum += e[nt]; }
#pragma unroll
        for (int d2 = 1; d2 < 16; d2 <<= 1)
          sum += __shfl_xor(sum, d2, 32);
        float inv = 1.0f / sum;
#pragma unroll
        for (int nt = 0; nt < 4; ++nt)
          Ph[m * 64 + nt * 16 + col] = (bf16)(e[nt] * inv);
      }
    }
  }
  __syncthreads();

  if (w < NH) {
    const int h = w;
    const bf16* Ph = sP + h * 4096;
    const bf16* Vh = sV + h * 2048;
    // ---- ctx = P @ V : M=64, N=32, K=64 (2 K-steps) ----
    v8f accC[4][2] = {};
#pragma unroll
    for (int ks = 0; ks < 2; ++ks) {
      v16bf pa[4], vb[2];
#pragma unroll
      for (int mt = 0; mt < 4; ++mt)
        pa[mt] = load_a_frag(Ph + mt * 16 * 64 + ks * 32, 64);
#pragma unroll
      for (int dt = 0; dt < 2; ++dt)
        vb[dt] = load_b_frag(Vh + dt * 16 * 64 + ks * 32, 64);
#pragma unroll
      for (int mt = 0; mt < 4; ++mt)
#pragma unroll
        for (int dt = 0; dt < 2; ++dt)
          accC[mt][dt] = wmma_bf16(pa[mt], vb[dt], accC[mt][dt]);
    }
#pragma unroll
    for (int mt = 0; mt < 4; ++mt)
#pragma unroll
      for (int dt = 0; dt < 2; ++dt)
#pragma unroll
        for (int r = 0; r < 8; ++r) {
          int m  = mt * 16 + hi * 8 + r;
          int ch = h * 32 + dt * 16 + col;
          sC[m * CD + ch] = (bf16)accC[mt][dt][r];
        }
  }
  __syncthreads();

  // ---- out = ctx @ w_proj + bias : 4 x 12 tiles, 6 per wave, K=192 ----
#pragma unroll
  for (int i = 0; i < 6; ++i) {
    int t  = w + i * 8;
    int mt = t & 3, ct = t >> 2;
    v8f acc = {};
#pragma unroll
    for (int ks = 0; ks < 6; ++ks) {
      v16bf a   = load_a_frag(sC + mt * 16 * CD + ks * 32, CD);
      v16bf bfr = load_b_frag(wpt + ct * 16 * CD + ks * 32, CD);  // L2-resident
      acc = wmma_bf16(a, bfr, acc);
    }
#pragma unroll
    for (int r = 0; r < 8; ++r) {
      int m = mt * 16 + hi * 8 + r;
      if (m < NT) {
        int c = ct * 16 + col;
        out[((size_t)b * NT + m) * CD + c] = acc[r] + bias[c];
      }
    }
  }
}

// ---------------------------------------------------------------------------
extern "C" void kernel_launch(void* const* d_in, const int* in_sizes, int n_in,
                              void* d_out, int out_size, void* d_ws, size_t ws_size,
                              hipStream_t stream) {
  const float* x     = (const float*)d_in[0];
  const float* mask  = (const float*)d_in[1];
  const float* wqkv  = (const float*)d_in[2];
  const float* wproj = (const float*)d_in[3];
  const float* bias  = (const float*)d_in[4];
  float*       out   = (float*)d_out;

  // workspace (bf16 elems):
  //   Q | K (each [4096][6][64][32]) | Vt ([4096][6][32][64]) | wqkvT | wpt
  const size_t nPad = (size_t)NB * NH * 64 * HD;   // 50,331,648 elems
  bf16* Qw    = (bf16*)d_ws;
  bf16* Kw    = Qw + nPad;
  bf16* Vt    = Kw + nPad;
  bf16* wqkvT = Vt + nPad;
  bf16* wpt   = wqkvT + (size_t)3 * CD * CD;

  k_prep<<<dim3((4 * CD * CD + 255) / 256), dim3(256), 0, stream>>>(wqkv, wproj, wqkvT, wpt);
  k_zero_pads<<<dim3(2048), dim3(256), 0, stream>>>(Qw, Kw, Vt);
  k_qkv<<<dim3(9, 3136), dim3(256), 0, stream>>>(x, wqkvT, Qw, Kw, Vt);
  k_attn_proj<<<dim3(NB), dim3(256), 0, stream>>>(Qw, Kw, Vt, mask, wpt, bias, out);
}
